// GTSModel_43739946942871
// MI455X (gfx1250) — compile-verified, matrix operand order checked by reading, not run on hardware
//
#include <hip/hip_runtime.h>
#include <math.h>

// ---------------------------------------------------------------------------
// Types / helpers
// ---------------------------------------------------------------------------
typedef __attribute__((ext_vector_type(16))) __bf16 v16bf;
typedef __attribute__((ext_vector_type(8)))  float  v8f;

union BFrag { unsigned short u[16]; v16bf v; };
union V4U16 { uint4 q; unsigned short s[8]; };

__device__ __forceinline__ unsigned short f2bf(float f) {
  unsigned u = __float_as_uint(f);
  u += 0x7fffu + ((u >> 16) & 1u);          // round-to-nearest-even
  return (unsigned short)(u >> 16);
}
__device__ __forceinline__ float bf2f(unsigned short h) {
  return __uint_as_float(((unsigned)h) << 16);
}

#define TM 128
#define TN 64
#define TK 32
#define LDSP 48          // padded row stride (elements); 96B = 6x16B, keeps uint4 stores aligned

#define ACT_NONE 0
#define ACT_RELU 1
#define ACT_SIG  2
#define ACT_TANH 3

// ---------------------------------------------------------------------------
// CDNA5 TDM probe (not launched; proves tensor_load_to_lds / s_wait_tensorcnt
// lower on this toolchain before we wire TDM into the GEMM staging).
// ---------------------------------------------------------------------------
typedef __attribute__((ext_vector_type(4))) unsigned int v4u32;
typedef __attribute__((ext_vector_type(8))) int          v8i32;
typedef __attribute__((ext_vector_type(4))) int          v4i32;

__global__ void tdm_probe_k()
{
  v4u32 g0 = {0, 0, 0, 0};
  v8i32 g1 = {0, 0, 0, 0, 0, 0, 0, 0};
  v4i32 g2 = {0, 0, 0, 0};
  v4i32 g3 = {0, 0, 0, 0};
#if defined(__clang_major__) && __clang_major__ >= 23
  v8i32 g4 = {0, 0, 0, 0, 0, 0, 0, 0};
  __builtin_amdgcn_tensor_load_to_lds(g0, g1, g2, g3, g4, 0);
#else
  __builtin_amdgcn_tensor_load_to_lds(g0, g1, g2, g3, 0);
#endif
  __builtin_amdgcn_s_wait_tensorcnt(0);
}

// ---------------------------------------------------------------------------
// Generic WMMA bf16 GEMM:
//   C = act( alpha * A@B + beta * Cin + bias )
// A: (M x K) bf16 row-major (lda), batch stride sA (elements, 0 = shared)
// B: (K x N) bf16 row-major (ldb), batch stride sB
// Cin: optional bf16 (ldcin, sCin); C: f32 or bf16 (ldc, sC)
// grid = (ceil(N/64), ceil(M/128), batch), block = 256 (8 waves)
// Each wave computes a 32x32 output tile (4 x v_wmma_f32_16x16x32_bf16 / K-step).
// Fast staging path: uint4 global loads -> uint4/scatter LDS stores, taken when
// the tile is interior and the operand row stride is 8-element aligned.
// ---------------------------------------------------------------------------
__global__ __launch_bounds__(256)
void wmma_gemm(const unsigned short* __restrict__ A, long long sA, int lda,
               const unsigned short* __restrict__ B, long long sB, int ldb,
               const unsigned short* __restrict__ Cin, long long sCin, int ldcin,
               void* __restrict__ C, long long sC, int ldc,
               const float* __restrict__ bias,
               int M, int Nn, int K,
               float alpha, float beta, int act, int out_bf16)
{
  __shared__ unsigned short As[TM][LDSP];
  __shared__ unsigned short Bs[TN][LDSP];

  const int bz = blockIdx.z;
  A += (long long)bz * sA;
  B += (long long)bz * sB;
  const unsigned short* Ci = Cin ? (Cin + (long long)bz * sCin) : nullptr;
  unsigned short* Cb16 = nullptr; float* Cb32 = nullptr;
  if (out_bf16) Cb16 = (unsigned short*)C + (long long)bz * sC;
  else          Cb32 = (float*)C + (long long)bz * sC;

  const int m0 = blockIdx.y * TM;
  const int n0 = blockIdx.x * TN;
  const int tid  = threadIdx.x;
  const int lane = tid & 31;
  const int wid  = tid >> 5;     // 0..7
  const int wm   = wid >> 1;     // 0..3 -> M offset wm*32
  const int wn   = wid & 1;      // 0..1 -> N offset wn*32
  const int row  = lane & 15;
  const int kh   = lane >> 4;

  const bool alignA = ((lda & 7) == 0) && (m0 + TM <= M);
  const bool alignB = ((ldb & 7) == 0) && (n0 + TN <= Nn);

  v8f acc[2][2];
#pragma unroll
  for (int mi = 0; mi < 2; mi++)
#pragma unroll
    for (int ni = 0; ni < 2; ni++)
      acc[mi][ni] = (v8f){0.f,0.f,0.f,0.f,0.f,0.f,0.f,0.f};

  for (int k0 = 0; k0 < K; k0 += TK) {
    const bool fullK = (k0 + TK <= K);

    // ---- stage A tile (TM x TK) ----
    if (fullK && alignA) {
      // 128 rows x 64B: 2 x uint4 per thread, all 16B aligned
#pragma unroll
      for (int i = 0; i < 2; i++) {
        int idx = tid + (i << 8);
        int r = idx >> 2, c = (idx & 3) << 3;
        V4U16 t;
        t.q = *(const uint4*)(A + (long long)(m0 + r) * lda + k0 + c);
        *(uint4*)&As[r][c] = t.q;
      }
    } else {
      for (int idx = tid; idx < TM * TK; idx += 256) {
        int r = idx / TK, c = idx - r * TK;
        int gm = m0 + r, gk = k0 + c;
        As[r][c] = (gm < M && gk < K) ? A[(long long)gm * lda + gk] : (unsigned short)0;
      }
    }
    // ---- stage B tile transposed: Bs[n][k] ----
    if (fullK && alignB) {
      // 32 rows x 128B: 1 x uint4 per thread, scatter-transpose into LDS
      int r = tid >> 3, c = (tid & 7) << 3;
      V4U16 t;
      t.q = *(const uint4*)(B + (long long)(k0 + r) * ldb + n0 + c);
#pragma unroll
      for (int e = 0; e < 8; e++) Bs[c + e][r] = t.s[e];
    } else {
      for (int idx = tid; idx < TN * TK; idx += 256) {
        int n = idx / TK, c = idx - n * TK;
        int gk = k0 + c, gn = n0 + n;
        Bs[n][c] = (gk < K && gn < Nn) ? B[(long long)gk * ldb + gn] : (unsigned short)0;
      }
    }
    // gfx1250 prefetch of next K tile while LDS fills
    if (k0 + TK < K) {
      int pm = m0 + (tid & 127); if (pm >= M) pm = M - 1;
      __builtin_prefetch((const void*)&A[(long long)pm * lda + (k0 + TK)], 0, 1);
      int pk = k0 + TK + (tid & 31); if (pk >= K) pk = K - 1;
      __builtin_prefetch((const void*)&B[(long long)pk * ldb + n0], 0, 1);
    }
    __syncthreads();

    // ---- fragments (ISA 7.12.2: 16-bit A 16x32 lane/VGPR layout; B mirrored) ----
    BFrag af[2], bf[2];
#pragma unroll
    for (int e = 0; e < 16; e++) {
      int kk = ((e >> 3) << 4) + (kh << 3) + (e & 7);
      af[0].u[e] = As[wm * 32 + row][kk];
      af[1].u[e] = As[wm * 32 + 16 + row][kk];
      bf[0].u[e] = Bs[wn * 32 + row][kk];
      bf[1].u[e] = Bs[wn * 32 + 16 + row][kk];
    }
#pragma unroll
    for (int mi = 0; mi < 2; mi++)
#pragma unroll
      for (int ni = 0; ni < 2; ni++)
        acc[mi][ni] = __builtin_amdgcn_wmma_f32_16x16x32_bf16(
            false, af[mi].v, false, bf[ni].v, (short)0, acc[mi][ni], false, false);
    __syncthreads();
  }

  // ---- epilogue (D layout: M = vgpr + 8*(lane>>4), N = lane&15) ----
  const int ncol = lane & 15;
  auto emit = [&](float av, int mm, int nn) {
    float f = alpha * av;
    if (Ci && beta != 0.f) f += beta * bf2f(Ci[(long long)mm * ldcin + nn]);
    if (bias) f += bias[nn];
    if (act == ACT_RELU)      f = fmaxf(f, 0.f);
    else if (act == ACT_SIG)  f = 1.f / (1.f + expf(-f));
    else if (act == ACT_TANH) f = tanhf(f);
    if (Cb16) Cb16[(long long)mm * ldc + nn] = f2bf(f);
    else      Cb32[(long long)mm * ldc + nn] = f;
  };
#pragma unroll
  for (int mi = 0; mi < 2; mi++) {
#pragma unroll
    for (int v = 0; v < 8; v++) {
      int mm = m0 + wm * 32 + mi * 16 + v + 8 * kh;
      if (mm >= M) continue;
#pragma unroll
      for (int ni = 0; ni < 2; ni++) {
        int nn = n0 + wn * 32 + ni * 16 + ncol;
        if (nn < Nn) emit(acc[mi][ni][v], mm, nn);
      }
    }
  }
}

// ---------------------------------------------------------------------------
// Feature extractor kernels
// ---------------------------------------------------------------------------
__global__ void conv1_k(const float* __restrict__ nf, const float* __restrict__ w,
                        const float* __restrict__ bias, float* __restrict__ out,
                        float* __restrict__ stats)
{
  int n = blockIdx.x, co = blockIdx.y, tid = threadIdx.x;
  __shared__ float wk[10];
  __shared__ float r1[256], r2[256];
  if (tid < 10) wk[tid] = w[co * 10 + tid];
  __syncthreads();
  float s1 = 0.f, s2 = 0.f, bb = bias[co];
  for (int t = tid; t < 1991; t += 256) {
    float a = bb;
#pragma unroll
    for (int k = 0; k < 10; k++) a += nf[(t + k) * 500 + n] * wk[k];
    a = fmaxf(a, 0.f);
    out[((long long)n * 8 + co) * 1991 + t] = a;
    s1 += a; s2 += a * a;
  }
  r1[tid] = s1; r2[tid] = s2; __syncthreads();
  for (int st = 128; st > 0; st >>= 1) {
    if (tid < st) { r1[tid] += r1[tid + st]; r2[tid] += r2[tid + st]; }
    __syncthreads();
  }
  if (tid == 0) { atomicAdd(&stats[co * 2], r1[0]); atomicAdd(&stats[co * 2 + 1], r2[0]); }
}

__global__ void conv2_k(const float* __restrict__ x, const float* __restrict__ w,
                        const float* __restrict__ bias, float* __restrict__ out,
                        float* __restrict__ stats)
{
  int n = blockIdx.x, co = blockIdx.y, tid = threadIdx.x;
  __shared__ float wk[80];
  __shared__ float r1[256], r2[256];
  if (tid < 80) wk[tid] = w[co * 80 + tid];
  __syncthreads();
  float s1 = 0.f, s2 = 0.f, bb = bias[co];
  for (int t = tid; t < 1982; t += 256) {
    float a = bb;
    for (int ci = 0; ci < 8; ci++) {
      const float* xp = x + ((long long)n * 8 + ci) * 1991 + t;
#pragma unroll
      for (int k = 0; k < 10; k++) a += xp[k] * wk[ci * 10 + k];
    }
    a = fmaxf(a, 0.f);
    out[((long long)n * 16 + co) * 1982 + t] = a;
    s1 += a; s2 += a * a;
  }
  r1[tid] = s1; r2[tid] = s2; __syncthreads();
  for (int st = 128; st > 0; st >>= 1) {
    if (tid < st) { r1[tid] += r1[tid + st]; r2[tid] += r2[tid + st]; }
    __syncthreads();
  }
  if (tid == 0) { atomicAdd(&stats[co * 2], r1[0]); atomicAdd(&stats[co * 2 + 1], r2[0]); }
}

__global__ void bn_apply(float* __restrict__ x, const float* __restrict__ stats,
                         const float* __restrict__ g, const float* __restrict__ b,
                         int C, int T, long long total, float invcnt,
                         unsigned short* __restrict__ obf)
{
  long long i = blockIdx.x * 256LL + threadIdx.x;
  if (i >= total) return;
  int ch = (int)((i / T) % C);
  float m   = stats[ch * 2] * invcnt;
  float var = stats[ch * 2 + 1] * invcnt - m * m;
  float v = (x[i] - m) * rsqrtf(var + 1e-5f) * g[ch] + b[ch];
  x[i] = v;
  if (obf) obf[i] = f2bf(v);
}

__global__ void bn3_k(float* __restrict__ x, const float* __restrict__ g,
                      const float* __restrict__ b, unsigned short* __restrict__ xbf)
{
  int e = threadIdx.x;
  if (e >= 100) return;
  float s1 = 0.f, s2 = 0.f;
  for (int n = 0; n < 500; n++) { float v = x[n * 100 + e]; s1 += v; s2 += v * v; }
  float m = s1 / 500.f, var = s2 / 500.f - m * m, inv = rsqrtf(var + 1e-5f);
  for (int n = 0; n < 500; n++) {
    float v = (x[n * 100 + e] - m) * inv * g[e] + b[e];
    x[n * 100 + e] = v;
    xbf[n * 100 + e] = f2bf(v);
  }
}

// ---------------------------------------------------------------------------
// Pairwise / adjacency kernels
// ---------------------------------------------------------------------------
__global__ void pair_k(const float* __restrict__ Ap, const float* __restrict__ Bp,
                       const float* __restrict__ catw, const float* __restrict__ catb,
                       const float* __restrict__ gum, float* __restrict__ amat)
{
  int idx = blockIdx.x * 256 + threadIdx.x;
  if (idx >= 250000) return;
  int i = idx / 500, j = idx - i * 500;
  float l0 = catb[0], l1 = catb[1];
  const float* aj = Ap + j * 100;
  const float* bi = Bp + i * 100;
  for (int k = 0; k < 100; k++) {
    float hh = fmaxf(aj[k] + bi[k], 0.f);
    l0 += hh * catw[2 * k]; l1 += hh * catw[2 * k + 1];
  }
  float g0 = -logf(-logf(gum[2 * idx] + 1e-20f) + 1e-20f);
  float g1 = -logf(-logf(gum[2 * idx + 1] + 1e-20f) + 1e-20f);
  float adjv = ((l0 + g0) >= (l1 + g1)) ? 1.f : 0.f;   // straight-through hard sample, class 0
  if (i == j) adjv = 1.f;                               // adj*(1-I) then +I
  amat[idx] = adjv;
}

__global__ void row_sum_k(const float* __restrict__ a, float* __restrict__ rs)
{
  int j = blockIdx.x, tid = threadIdx.x;
  __shared__ float red[256];
  float s = 0.f;
  for (int i = tid; i < 500; i += 256) s += a[j * 500 + i];
  red[tid] = s; __syncthreads();
  for (int st = 128; st > 0; st >>= 1) {
    if (tid < st) red[tid] += red[tid + st];
    __syncthreads();
  }
  if (tid == 0) rs[j] = red[0];
}

// support = (a/rowsum).T, stored bf16 with ld=512 (16B-aligned rows)
__global__ void supp_k(const float* __restrict__ amat, const float* __restrict__ rs,
                       unsigned short* __restrict__ sb)
{
  int idx = blockIdx.x * 256 + threadIdx.x;
  if (idx >= 250000) return;
  int i = idx / 500, j = idx - i * 500;
  sb[i * 512 + j] = f2bf(amat[j * 500 + i] / rs[j]);
}

// ---------------------------------------------------------------------------
// Param prep / elementwise recurrent kernels
// ---------------------------------------------------------------------------
// De-interleave stacked-K gate weights: dst rows j*dp + k (zero for k in [d,dp))
__global__ void reorder_w_k(const float* __restrict__ src, unsigned short* __restrict__ dst,
                            int d, int dp, int outc, int total)
{
  int i = blockIdx.x * 256 + threadIdx.x;
  if (i >= total) return;
  int o = i % outc, rrow = i / outc;
  int j = rrow / dp, k = rrow - j * dp;
  dst[i] = (k < d) ? f2bf(src[(k * 3 + j) * outc + o]) : (unsigned short)0;
}

__global__ void cvt_bf_k(const float* __restrict__ s, unsigned short* __restrict__ t, long long n)
{
  long long i = blockIdx.x * 256LL + threadIdx.x;
  if (i < n) t[i] = f2bf(s[i]);
}

__global__ void fillzero_k(float* __restrict__ p, long long n)
{
  long long i = blockIdx.x * 256LL + threadIdx.x;
  if (i < n) p[i] = 0.f;
}

// z = concat(x, r?r*h:h) written as bf16 into Zcat[:, 0:d] (row length ld)
__global__ void build_z_k(const float* __restrict__ x, int din, const float* __restrict__ h,
                          const float* __restrict__ r, unsigned short* __restrict__ z,
                          int ld, int d, long long total)
{
  long long i = blockIdx.x * 256LL + threadIdx.x;
  if (i >= total) return;
  long long rowi = i / d; int c = (int)(i - rowi * d);
  float v;
  if (c < din) v = x[rowi * din + c];
  else {
    int k = c - din;
    float hv = h[rowi * 128 + k];
    v = r ? r[rowi * 256 + k] * hv : hv;
  }
  z[rowi * ld + c] = f2bf(v);
}

__global__ void gru_update_k(float* __restrict__ h, const float* __restrict__ ru,
                             const float* __restrict__ c, long long n)
{
  long long i = blockIdx.x * 256LL + threadIdx.x;
  if (i >= n) return;
  long long rowi = i >> 7; int k = (int)(i & 127);
  float u = ru[rowi * 256 + 128 + k];
  h[i] = u * h[i] + (1.f - u) * c[i];
}

__global__ void proj_k(const float* __restrict__ h1, const float* __restrict__ pw,
                       const float* __restrict__ pb, float* __restrict__ out_t,
                       float* __restrict__ go)
{
  int idx = blockIdx.x * 256 + threadIdx.x;
  if (idx >= 32000) return;
  float acc = pb[0];
  const float* hp = h1 + (long long)idx * 128;
  for (int k = 0; k < 128; k++) acc += hp[k] * pw[k];
  out_t[idx] = acc;
  go[idx] = acc;
}

// ---------------------------------------------------------------------------
// Host-side helpers
// ---------------------------------------------------------------------------
static void launch_gemm(hipStream_t s,
                        const unsigned short* A, long long sA, int lda,
                        const unsigned short* B, long long sB, int ldb,
                        const unsigned short* Cin, long long sCin, int ldcin,
                        void* C, long long sC, int ldc,
                        const float* bias, int M, int N, int K,
                        float alpha, float beta, int act, int obf, int batch)
{
  dim3 g((N + TN - 1) / TN, (M + TM - 1) / TM, batch);
  wmma_gemm<<<g, 256, 0, s>>>(A, sA, lda, B, sB, ldb, Cin, sCin, ldcin,
                              C, sC, ldc, bias, M, N, K, alpha, beta, act, obf);
}

static void run_dcgru(hipStream_t s, const float* x, int din, float* h,
                      const unsigned short* supp,
                      const unsigned short* Wru, const float* bru,
                      const unsigned short* Wc,  const float* bc,
                      unsigned short* zcat, float* ru, float* cbuf)
{
  const int d  = din + 128;
  const int dp = (d + 7) & ~7;        // 8-element aligned column blocks
  const int ld = 3 * dp;              // Zcat leading dimension
  const long long tot = 32000LL * d;
  const int zb = (int)((tot + 255) / 256);
  const long long sZ = 500LL * ld;    // per-batch stride inside Zcat

  // --- r/u gate ---
  build_z_k<<<zb, 256, 0, s>>>(x, din, h, nullptr, zcat, ld, d, tot);
  // z1 = S @ z   (batched over B=64)
  launch_gemm(s, supp, 0, 512, zcat, sZ, ld, nullptr, 0, 0,
              (void*)(zcat + dp), sZ, ld, nullptr, 500, d, 500, 1.f, 0.f, ACT_NONE, 1, 64);
  // z2 = 2*S@z1 - z0
  launch_gemm(s, supp, 0, 512, zcat + dp, sZ, ld, zcat, sZ, ld,
              (void*)(zcat + 2 * dp), sZ, ld, nullptr, 500, d, 500, 2.f, -1.f, ACT_NONE, 1, 64);
  // ru = sigmoid(Zcat @ Wru + bru)
  launch_gemm(s, zcat, 0, ld, Wru, 0, 256, nullptr, 0, 0,
              (void*)ru, 0, 256, bru, 32000, 256, ld, 1.f, 0.f, ACT_SIG, 0, 1);

  // --- candidate ---
  build_z_k<<<zb, 256, 0, s>>>(x, din, h, ru, zcat, ld, d, tot);
  launch_gemm(s, supp, 0, 512, zcat, sZ, ld, nullptr, 0, 0,
              (void*)(zcat + dp), sZ, ld, nullptr, 500, d, 500, 1.f, 0.f, ACT_NONE, 1, 64);
  launch_gemm(s, supp, 0, 512, zcat + dp, sZ, ld, zcat, sZ, ld,
              (void*)(zcat + 2 * dp), sZ, ld, nullptr, 500, d, 500, 2.f, -1.f, ACT_NONE, 1, 64);
  launch_gemm(s, zcat, 0, ld, Wc, 0, 128, nullptr, 0, 0,
              (void*)cbuf, 0, 128, bc, 32000, 128, ld, 1.f, 0.f, ACT_TANH, 0, 1);

  gru_update_k<<<(32000 * 128 + 255) / 256, 256, 0, s>>>(h, ru, cbuf, 32000LL * 128);
}

// ---------------------------------------------------------------------------
// Entry point
// ---------------------------------------------------------------------------
extern "C" void kernel_launch(void* const* d_in, const int* in_sizes, int n_in,
                              void* d_out, int out_size, void* d_ws, size_t ws_size,
                              hipStream_t stream)
{
  (void)in_sizes; (void)n_in; (void)out_size; (void)ws_size;

  const float* nf     = (const float*)d_in[0];   // (L=2000, N=500)
  const float* inputs = (const float*)d_in[1];   // (12, 64, 1000)
  const float* gum    = (const float*)d_in[2];   // (250000, 2)
  const float* const* P = (const float* const*)(d_in + 3);   // params, insertion order

  const float* conv1w = P[0];  const float* conv1b = P[1];
  const float* bn1g   = P[2];  const float* bn1b   = P[3];
  const float* conv2w = P[4];  const float* conv2b = P[5];
  const float* bn2g   = P[6];  const float* bn2b   = P[7];
  const float* fcw    = P[8];  const float* fcb    = P[9];
  const float* bn3g   = P[10]; const float* bn3b   = P[11];
  const float* fcow   = P[12]; const float* fcob   = P[13];
  const float* catw   = P[14]; const float* catb   = P[15];
  const float* projw  = P[16]; const float* projb  = P[17];
  const float* Wru_f[4] = { P[18], P[22], P[26], P[30] };   // enc0, enc1, dec0, dec1
  const float* bru_f[4] = { P[19], P[23], P[27], P[31] };
  const float* Wc_f [4] = { P[20], P[24], P[28], P[32] };
  const float* bc_f [4] = { P[21], P[25], P[29], P[33] };
  const int dcell[4]  = { 130, 256, 129, 256 };   // din + U
  const int dpcell[4] = { 136, 256, 136, 256 };   // padded to 8

  // ---- workspace carve-up ----
  char* W = (char*)d_ws;
  size_t off = 0;
  auto take = [&](size_t bytes) { size_t r = off; off += (bytes + 255) & ~(size_t)255; return r; };
  size_t o_h0   = take(32000ULL * 128 * 4);   // hidden layer0 (f32)
  size_t o_h1   = take(32000ULL * 128 * 4);   // hidden layer1
  size_t o_ru   = take(32000ULL * 256 * 4);   // ru gate (f32)
  size_t o_c    = take(32000ULL * 128 * 4);   // candidate (f32)
  size_t o_z    = take(32000ULL * 768 * 2);   // Zcat (bf16, max ld=768)
  size_t o_go   = take(32000ULL * 4);
  size_t o_supp = take(500ULL * 512 * 2);     // support bf16, ld=512
  size_t o_amat = take(250000ULL * 4);        // adj + I (f32)
  size_t o_rs   = take(512 * 4);
  size_t o_st   = take(64 * 4);               // bn1/bn2 stats
  size_t o_fcw  = take(3171200ULL * 2);       // fc_w bf16
  size_t o_fco  = take(20000ULL * 2);         // fc_out_w bf16
  size_t o_xe   = take(50000ULL * 4);         // x embedding f32 (500x100)
  size_t o_xeb  = take(50000ULL * 2);         // x embedding bf16
  size_t o_Ap   = take(50000ULL * 4);
  size_t o_Bp   = take(50000ULL * 4);
  size_t o_wru[4], o_wc[4];
  for (int i = 0; i < 4; i++) {
    o_wru[i] = take((size_t)3 * dpcell[i] * 256 * 2);
    o_wc[i]  = take((size_t)3 * dpcell[i] * 128 * 2);
  }

  float* h0   = (float*)(W + o_h0);
  float* h1   = (float*)(W + o_h1);
  float* ru   = (float*)(W + o_ru);
  float* cbuf = (float*)(W + o_c);
  unsigned short* zcat = (unsigned short*)(W + o_z);
  float* go   = (float*)(W + o_go);
  unsigned short* suppb = (unsigned short*)(W + o_supp);
  float* amat = (float*)(W + o_amat);
  float* rs   = (float*)(W + o_rs);
  float* stats = (float*)(W + o_st);
  unsigned short* fcwb = (unsigned short*)(W + o_fcw);
  unsigned short* fcob2 = (unsigned short*)(W + o_fco);
  float* xemb = (float*)(W + o_xe);
  unsigned short* xembb = (unsigned short*)(W + o_xeb);
  float* Ap = (float*)(W + o_Ap);
  float* Bp = (float*)(W + o_Bp);

  // Feature-stage aliases (dead by the time the recurrent stage starts):
  float* conv1out = (float*)(W + o_h0);           // 500*8*1991*4  = 31.9 MB over h0+h1
  float* conv2out = (float*)(W + o_ru);           // 500*16*1982*4 = 63.4 MB over ru+c+zcat
  unsigned short* xfeatb = (unsigned short*)(W + o_h0);   // 500*31712*2 = 31.7 MB (after conv1 dead)

  // =========================== feature extractor ===========================
  fillzero_k<<<1, 64, 0, stream>>>(stats, 64);
  conv1_k<<<dim3(500, 8), 256, 0, stream>>>(nf, conv1w, conv1b, conv1out, stats);
  bn_apply<<<(int)((500LL * 8 * 1991 + 255) / 256), 256, 0, stream>>>(
      conv1out, stats, bn1g, bn1b, 8, 1991, 500LL * 8 * 1991, 1.f / (500.f * 1991.f), nullptr);
  conv2_k<<<dim3(500, 16), 256, 0, stream>>>(conv1out, conv2w, conv2b, conv2out, stats + 16);
  bn_apply<<<(int)((500LL * 16 * 1982 + 255) / 256), 256, 0, stream>>>(
      conv2out, stats + 16, bn2g, bn2b, 16, 1982, 500LL * 16 * 1982, 1.f / (500.f * 1982.f), xfeatb);

  cvt_bf_k<<<(int)((3171200 + 255) / 256), 256, 0, stream>>>(fcw, fcwb, 3171200);
  // xemb = relu(xfeat @ fc_w + fc_b) : 500 x 31712 x 100 WMMA GEMM
  launch_gemm(stream, xfeatb, 0, 31712, fcwb, 0, 100, nullptr, 0, 0,
              (void*)xemb, 0, 100, fcb, 500, 100, 31712, 1.f, 0.f, ACT_RELU, 0, 1);
  bn3_k<<<1, 128, 0, stream>>>(xemb, bn3g, bn3b, xembb);

  // pairwise MLP collapsed: Ap = x@W_top, Bp = x@W_bot + b
  cvt_bf_k<<<(20000 + 255) / 256, 256, 0, stream>>>(fcow, fcob2, 20000);
  launch_gemm(stream, xembb, 0, 100, fcob2, 0, 100, nullptr, 0, 0,
              (void*)Ap, 0, 100, nullptr, 500, 100, 100, 1.f, 0.f, ACT_NONE, 0, 1);
  launch_gemm(stream, xembb, 0, 100, fcob2 + 100 * 100, 0, 100, nullptr, 0, 0,
              (void*)Bp, 0, 100, fcob, 500, 100, 100, 1.f, 0.f, ACT_NONE, 0, 1);
  pair_k<<<(250000 + 255) / 256, 256, 0, stream>>>(Ap, Bp, catw, catb, gum, amat);
  row_sum_k<<<500, 256, 0, stream>>>(amat, rs);
  supp_k<<<(250000 + 255) / 256, 256, 0, stream>>>(amat, rs, suppb);

  // prep de-interleaved bf16 gate weights (rows j*dp + k, zero-padded)
  for (int i = 0; i < 4; i++) {
    int tot_ru = 3 * dpcell[i] * 256, tot_c = 3 * dpcell[i] * 128;
    reorder_w_k<<<(tot_ru + 255) / 256, 256, 0, stream>>>(
        Wru_f[i], (unsigned short*)(W + o_wru[i]), dcell[i], dpcell[i], 256, tot_ru);
    reorder_w_k<<<(tot_c + 255) / 256, 256, 0, stream>>>(
        Wc_f[i], (unsigned short*)(W + o_wc[i]), dcell[i], dpcell[i], 128, tot_c);
  }

  // =============================== encoder =================================
  // zero hidden states and the whole Zcat (pad columns must be 0)
  fillzero_k<<<(int)((2LL * 32000 * 128 + 255) / 256), 256, 0, stream>>>(h0, 2LL * 32000 * 128);
  fillzero_k<<<(int)((32000LL * 768 / 2 + 255) / 256), 256, 0, stream>>>(
      (float*)zcat, 32000LL * 768 / 2);
  float* hbuf[2] = { h0, h1 };
  for (int t = 0; t < 12; t++) {
    const float* x = inputs + (long long)t * 64 * 1000;   // (B, N, DIN=2) rows b*N+n
    int din = 2;
    for (int l = 0; l < 2; l++) {
      run_dcgru(stream, x, din, hbuf[l], suppb,
                (unsigned short*)(W + o_wru[l]), bru_f[l],
                (unsigned short*)(W + o_wc[l]),  bc_f[l],
                zcat, ru, cbuf);
      x = hbuf[l]; din = 128;
    }
  }

  // =============================== decoder =================================
  fillzero_k<<<(32000 + 255) / 256, 256, 0, stream>>>(go, 32000);
  float* out = (float*)d_out;                              // (HOR, B, N) f32
  for (int t = 0; t < 12; t++) {
    const float* x = go;
    int din = 1;
    for (int l = 0; l < 2; l++) {
      run_dcgru(stream, x, din, hbuf[l], suppb,
                (unsigned short*)(W + o_wru[2 + l]), bru_f[2 + l],
                (unsigned short*)(W + o_wc[2 + l]),  bc_f[2 + l],
                zcat, ru, cbuf);
      x = hbuf[l]; din = 128;
    }
    proj_k<<<(32000 + 255) / 256, 256, 0, stream>>>(h1, projw, projb, out + (long long)t * 32000, go);
  }
}
